// AnomalyAttention_59906203844839
// MI455X (gfx1250) — matrix operand order repaired and checked by reading, not compiled, not run on hardware
//
#include <hip/hip_runtime.h>
#include <math.h>

// AnomalyAttention forward for MI455X (gfx1250, wave32, WMMA).
// Outputs concatenated flat: V[16,512,8,64], series[16,8,512,512],
// prior[16,8,512,512], sigma_full[16,8,512,512].
//
// d_ws layout (requires ws_size >= 16 MB):
//   [0, 8MB)   Kh : f16 keys,  [B,H,L,E]  (dim-contiguous rows)
//   [8, 16MB)  Vt : f16 values transposed, [B,H,E,L] (seq-contiguous rows)

typedef __attribute__((ext_vector_type(16))) _Float16 v16h;
typedef __attribute__((ext_vector_type(8)))  _Float16 v8h;
typedef __attribute__((ext_vector_type(4)))  _Float16 v4h;
typedef __attribute__((ext_vector_type(8)))  float    v8f;

#define B_ 16
#define L_ 512
#define H_ 8
#define E_ 64
#define HE 512          // H_*E_ : f32 stride (floats) between sequence positions
#define ROWSTRIDE 516   // f32 LDS row pad: 516%64==4 -> 16 distinct banks
#define WBSTRIDE 520    // f16 LDS row pad (halves): 1040B rows, 16B-aligned

// ---------------------------------------------------------------------------
// Pre-pass 1: K (f32, [B,L,H,E]) -> Kh (f16, [B,H,L,E]). Pure streaming.
// ---------------------------------------------------------------------------
__global__ __launch_bounds__(256)
void cvt_k_kernel(const float* __restrict__ K, _Float16* __restrict__ Kh) {
  const size_t o = ((size_t)blockIdx.x * 256 + threadIdx.x) * 4;  // out linear
  const int    e  = (int)(o & 63);
  const size_t r  = o >> 6;          // (b*H+h)*L + s
  const int    s  = (int)(r & 511);
  const size_t bh = r >> 9;          // b*H + h
  const int    h  = (int)(bh & 7);
  const int    b  = (int)(bh >> 3);
  const float4 t = *(const float4*)(K + (((size_t)b * L_ + s) * H_ + h) * E_ + e);
  v4h q;
  q.x = (_Float16)t.x; q.y = (_Float16)t.y;
  q.z = (_Float16)t.z; q.w = (_Float16)t.w;
  *(v4h*)(Kh + o) = q;               // 8-byte packed store
}

// ---------------------------------------------------------------------------
// Pre-pass 2: V (f32, [B,L,H,E]) -> Vt (f16, [B,H,E,L]) via LDS 64x64 tiles.
// ---------------------------------------------------------------------------
__global__ __launch_bounds__(256)
void cvt_v_kernel(const float* __restrict__ V, _Float16* __restrict__ Vt) {
  __shared__ float tile[64][65];     // +1 pad: conflict-free column reads
  const int t  = threadIdx.x;
  const int s0 = blockIdx.x * 64;    // seq tile origin
  const int h  = blockIdx.y;
  const int b  = blockIdx.z;
  const float* src = V + (((size_t)b * L_ + s0) * H_ + h) * E_;
#pragma unroll
  for (int j = 0; j < 4; ++j) {
    const int flat = t + 256 * j;            // 0..1023
    const int sl = flat >> 4;                // 0..63
    const int e0 = (flat & 15) * 4;
    const float4 v4 = *(const float4*)(src + (size_t)sl * HE + e0);
    tile[sl][e0] = v4.x; tile[sl][e0 + 1] = v4.y;
    tile[sl][e0 + 2] = v4.z; tile[sl][e0 + 3] = v4.w;
  }
  __syncthreads();
  _Float16* dst = Vt + ((size_t)b * H_ + h) * E_ * L_;
#pragma unroll
  for (int j = 0; j < 4; ++j) {
    const int flat = t + 256 * j;            // 0..1023
    const int d   = flat >> 4;               // 0..63
    const int sl0 = (flat & 15) * 4;
    v4h q;
    q.x = (_Float16)tile[sl0][d];     q.y = (_Float16)tile[sl0 + 1][d];
    q.z = (_Float16)tile[sl0 + 2][d]; q.w = (_Float16)tile[sl0 + 3][d];
    *(v4h*)(dst + (size_t)d * L_ + s0 + sl0) = q;
  }
}

// ---------------------------------------------------------------------------
// Fused attention: one wave32 per (b, h, 16-row) tile.
//   Phase 1: S = (Q*scale) Kh^T (16x512), double-buffered K fragments -> LDS
//   Phase 2: row softmax (2 lanes/row, shfl_xor cross-half), float4 passes;
//            writes f32 series to global + packed-f16 weights to LDS
//   Phase 3: O = weights @ V, double-buffered; A frags = 2x ds_load_b128
// ---------------------------------------------------------------------------
__global__ __launch_bounds__(32)
void attn_kernel(const float* __restrict__ Q, const _Float16* __restrict__ Kh,
                 const _Float16* __restrict__ Vt, float* __restrict__ Vout,
                 float* __restrict__ series) {
  __shared__ float    sc[16 * ROWSTRIDE];  // 33 KB f32 score strip
  __shared__ _Float16 wb[16 * WBSTRIDE];   // 16.6 KB f16 normalized weights

  const int lane = threadIdx.x;   // single full wave: EXEC all ones for WMMA
  const int n    = lane & 15;     // row (A frag) / column (B,C,D frags)
  const int hs   = lane >> 4;     // half-select
  const int row0 = blockIdx.x * 16;
  const int h    = blockIdx.y;
  const int b    = blockIdx.z;
  const size_t bh = (size_t)b * H_ + h;

  // ---- A fragments for Q rows, scale 1/sqrt(E)=0.125 folded in (exact,
  //      power-of-two exponent shift before f16 conversion).
  //      A layout (matches contiguous K frags): elem s -> K=ko+hs*16+s
  v16h qa0, qa1;
  {
    const float* p = Q + ((size_t)b * L_ * H_ + h) * E_ + (size_t)(row0 + n) * HE;
#pragma unroll
    for (int i = 0; i < 8; ++i) {
      qa0[i]     = (_Float16)(p[      hs * 16 + i] * 0.125f);
      qa0[8 + i] = (_Float16)(p[8 +   hs * 16 + i] * 0.125f);
      qa1[i]     = (_Float16)(p[32 +  hs * 16 + i] * 0.125f);
      qa1[8 + i] = (_Float16)(p[40 +  hs * 16 + i] * 0.125f);
    }
  }

  const _Float16* krow = Kh + (bh * L_ + n) * E_ + hs * 16;

  // ---- Phase 1: scores -> LDS, software-pipelined K fragments --------------
  v16h b0 = *(const v16h*)krow;
  v16h b1 = *(const v16h*)(krow + 32);
#pragma unroll
  for (int ct = 0; ct < 31; ++ct) {
    const _Float16* np = krow + (size_t)(ct + 1) * (16 * E_);
    __builtin_prefetch(np + 16 * E_, 0, 3);        // tile ct+2
    const v16h nb0 = *(const v16h*)np;             // tile ct+1 fragments
    const v16h nb1 = *(const v16h*)(np + 32);
    v8f c = {};
    c = __builtin_amdgcn_wmma_f32_16x16x32_f16(false, qa0, false, b0,
                                               (short)0, c, false, false);
    c = __builtin_amdgcn_wmma_f32_16x16x32_f16(false, qa1, false, b1,
                                               (short)0, c, false, false);
#pragma unroll
    for (int r = 0; r < 8; ++r)
      sc[(r + 8 * hs) * ROWSTRIDE + ct * 16 + n] = c[r];
    b0 = nb0; b1 = nb1;
  }
  {
    v8f c = {};
    c = __builtin_amdgcn_wmma_f32_16x16x32_f16(false, qa0, false, b0,
                                               (short)0, c, false, false);
    c = __builtin_amdgcn_wmma_f32_16x16x32_f16(false, qa1, false, b1,
                                               (short)0, c, false, false);
#pragma unroll
    for (int r = 0; r < 8; ++r)
      sc[(r + 8 * hs) * ROWSTRIDE + 31 * 16 + n] = c[r];
  }
  // single-wave workgroup: in-wave dscnt waits order LDS RAW hazards

  // ---- Phase 2: softmax; series (f32, global) + weights (f16, LDS) ---------
  {
    float* rp = sc + n * ROWSTRIDE + hs * 256;  // lane pair per row, 256 each
    float mx = -3.4e38f;
#pragma unroll 4
    for (int c4 = 0; c4 < 64; ++c4) {
      const float4 t = *(const float4*)(rp + 4 * c4);
      mx = fmaxf(mx, fmaxf(fmaxf(t.x, t.y), fmaxf(t.z, t.w)));
    }
    mx = fmaxf(mx, __shfl_xor(mx, 16, 32));
    float sum = 0.f;
#pragma unroll 4
    for (int c4 = 0; c4 < 64; ++c4) {
      float4 t = *(const float4*)(rp + 4 * c4);
      t.x = __expf(t.x - mx); t.y = __expf(t.y - mx);
      t.z = __expf(t.z - mx); t.w = __expf(t.w - mx);
      *(float4*)(rp + 4 * c4) = t;
      sum += (t.x + t.y) + (t.z + t.w);
    }
    sum += __shfl_xor(sum, 16, 32);
    const float inv = 1.f / sum;
    float*    gp = series + (bh * L_ + row0 + n) * L_ + hs * 256;
    _Float16* wp = wb + n * WBSTRIDE + hs * 256;
#pragma unroll 4
    for (int c4 = 0; c4 < 64; ++c4) {
      float4 t = *(const float4*)(rp + 4 * c4);
      t.x *= inv; t.y *= inv; t.z *= inv; t.w *= inv;
      *(float4*)(gp + 4 * c4) = t;          // series output (128-bit stores)
      v4h q;
      q.x = (_Float16)t.x; q.y = (_Float16)t.y;
      q.z = (_Float16)t.z; q.w = (_Float16)t.w;
      *(v4h*)(wp + 4 * c4) = q;             // packed weights for phase 3
    }
  }

  // ---- Phase 3: O = weights[16x512] @ V[512x64], double-buffered -----------
  const _Float16* vtbase = Vt + bh * E_ * L_ + hs * 16;
  auto loadA = [&](int ko) -> v16h {
    const _Float16* p = wb + n * WBSTRIDE + ko;
    const v8h lo = *(const v8h*)(p + hs * 16);        // ds_load_b128
    const v8h hi = *(const v8h*)(p + 8 + hs * 16);   // ds_load_b128
    v16h a;
#pragma unroll
    for (int i = 0; i < 8; ++i) { a[i] = lo[i]; a[8 + i] = hi[i]; }
    return a;
  };

  v16h a = loadA(0);
  v16h bf[4];
#pragma unroll
  for (int dt = 0; dt < 4; ++dt)
    bf[dt] = *(const v16h*)(vtbase + (size_t)(dt * 16 + n) * L_);

  v8f acc[4] = {};
#pragma unroll
  for (int kt = 0; kt < 15; ++kt) {
    const int nko = (kt + 1) * 32;
    const v16h na = loadA(nko);
    v16h nb[4];
#pragma unroll
    for (int dt = 0; dt < 4; ++dt)
      nb[dt] = *(const v16h*)(vtbase + (size_t)(dt * 16 + n) * L_ + nko);
#pragma unroll
    for (int dt = 0; dt < 4; ++dt)
      acc[dt] = __builtin_amdgcn_wmma_f32_16x16x32_f16(false, a, false, bf[dt],
                                                       (short)0, acc[dt],
                                                       false, false);
    a = na;
#pragma unroll
    for (int dt = 0; dt < 4; ++dt) bf[dt] = nb[dt];
  }
#pragma unroll
  for (int dt = 0; dt < 4; ++dt)
    acc[dt] = __builtin_amdgcn_wmma_f32_16x16x32_f16(false, a, false, bf[dt],
                                                     (short)0, acc[dt],
                                                     false, false);

  // store V output: [b, row, h, dim]
  float* vo = Vout + (((size_t)b * L_ + row0) * H_ + h) * E_;
#pragma unroll
  for (int dt = 0; dt < 4; ++dt)
#pragma unroll
    for (int r = 0; r < 8; ++r)
      vo[(size_t)(r + 8 * hs) * HE + dt * 16 + n] = acc[dt][r];
}

// ---------------------------------------------------------------------------
// prior + sigma_full: pure streaming elementwise (268 MB of writes).
// ---------------------------------------------------------------------------
__global__ __launch_bounds__(128)
void prior_kernel(const float* __restrict__ sigma, float* __restrict__ prior,
                  float* __restrict__ sigf) {
  const int l = blockIdx.x, h = blockIdx.y, b = blockIdx.z;
  const float x = sigma[((size_t)b * L_ + l) * H_ + h];
  const float s  = 1.f / (1.f + __expf(-5.f * x)) + 1e-5f;   // sigmoid(5x)+eps
  const float sg = __expf(s * 1.0986122886681098f) - 1.f;    // 3^s - 1
  const float amp = 0.3989422804014327f / sg;                // 1/(sqrt(2pi)*sig)
  const float c2  = -0.5f / (sg * sg);

  const size_t base = (((size_t)b * H_ + h) * L_ + l) * L_;
  const int j0 = threadIdx.x * 4;

  float4 pv, sv;
  float d;
  d = (float)(l - j0);       pv.x = amp * __expf(d * d * c2);
  d = (float)(l - j0 - 1);   pv.y = amp * __expf(d * d * c2);
  d = (float)(l - j0 - 2);   pv.z = amp * __expf(d * d * c2);
  d = (float)(l - j0 - 3);   pv.w = amp * __expf(d * d * c2);
  sv.x = sg; sv.y = sg; sv.z = sg; sv.w = sg;

  *(float4*)(prior + base + j0) = pv;
  *(float4*)(sigf  + base + j0) = sv;
}

// ---------------------------------------------------------------------------
extern "C" void kernel_launch(void* const* d_in, const int* in_sizes, int n_in,
                              void* d_out, int out_size, void* d_ws, size_t ws_size,
                              hipStream_t stream) {
  const float* q  = (const float*)d_in[0];
  const float* k  = (const float*)d_in[1];
  const float* v  = (const float*)d_in[2];
  const float* sg = (const float*)d_in[3];
  // d_in[4] = attn_mask (unused: mask_flag=False)

  float* out    = (float*)d_out;
  float* Vout   = out;                                   //  4,194,304 floats
  float* series = out + (size_t)B_ * L_ * H_ * E_;       // 33,554,432 floats
  float* prior  = series + (size_t)B_ * H_ * L_ * L_;    // 33,554,432 floats
  float* sigf   = prior  + (size_t)B_ * H_ * L_ * L_;    // 33,554,432 floats

  const size_t nqkv = (size_t)B_ * L_ * H_ * E_;         // 4,194,304 elements
  _Float16* Kh = (_Float16*)d_ws;                        // 8 MB
  _Float16* Vt = Kh + nqkv;                              // 8 MB

  cvt_k_kernel<<<dim3((unsigned)(nqkv / 4 / 256)), 256, 0, stream>>>(k, Kh);
  cvt_v_kernel<<<dim3(L_ / 64, H_, B_), 256, 0, stream>>>(v, Vt);
  attn_kernel<<<dim3(L_ / 16, H_, B_), 32, 0, stream>>>(q, Kh, Vt, Vout, series);
  prior_kernel<<<dim3(L_, H_, B_), 128, 0, stream>>>(sg, prior, sigf);
}
